// TtShiftedWindowAttention_85899346104
// MI455X (gfx1250) — matrix operand order repaired and checked by reading, not compiled, not run on hardware
//
#include <hip/hip_runtime.h>
#include <hip/hip_bf16.h>

// CDNA5 / gfx1250 fused Swin shifted-window attention.
// One wave32 per 7x7 window; all GEMMs via v_wmma_f32_16x16x32_bf16.
// Loops deliberately kept rolled to stay under 256 VGPRs (no s_set_vgpr_msb).

typedef __attribute__((ext_vector_type(16))) __bf16  v16bf;
typedef __attribute__((ext_vector_type(8)))  float   v8f;
typedef __attribute__((ext_vector_type(4)))  unsigned int u32x4;
typedef __attribute__((ext_vector_type(2)))  unsigned int u32x2;
typedef __attribute__((ext_vector_type(4)))  float       f32x4;

#define NHEADS 3
#define HD     32
#define CDIM   96
#define SWIN   7
#define SSHIFT 3
#define HW     112
#define SPAD   64   // 49 tokens padded to 64 (4 M-tiles)

// Native bf16 convert (RNE): lowers to v_cvt_bf16_f32 / v_cvt_pk_bf16_f32.
__device__ __forceinline__ unsigned short f2bf(float f) {
    __bf16 h = (__bf16)f;
    union { __bf16 b; unsigned short u; } c;
    c.b = h;
    return c.u;
}
__device__ __forceinline__ unsigned pk2bf(float lo, float hi) {
    return (unsigned)f2bf(lo) | ((unsigned)f2bf(hi) << 16);
}

union FragU { u32x4 q[2]; v16bf v; };

// A-matrix fragment (16x32 bf16), row-major source, per ISA layout:
// lane L holds row M = L%16; elems 0..7 -> K = kOff + half*8 + e,
// elems 8..15 -> K = kOff + 16 + half*8 + (e-8). Two 16B loads.
__device__ __forceinline__ v16bf ldA(const unsigned short* base, int row,
                                     int stride, int kOff, int half) {
    const unsigned short* p = base + row * stride + kOff + half * 8;
    FragU u;
    u.q[0] = *(const u32x4*)(p);
    u.q[1] = *(const u32x4*)(p + 16);
    return u.v;
}

// B-matrix fragment (32x16 bf16) from B^T ([N,K] row-major) source:
// lane L holds col N; elems 0..7 -> K = kOff + half*8 + e,
// elems 8..15 -> K = kOff + 16 + half*8 + (e-8). Two 16B loads.
__device__ __forceinline__ v16bf ldB(const unsigned short* rowp, int kOff, int half) {
    const unsigned short* p = rowp + kOff + half * 8;
    FragU u;
    u.q[0] = *(const u32x4*)(p);
    u.q[1] = *(const u32x4*)(p + 16);
    return u.v;
}

__device__ __forceinline__ v8f wmma_bf16(v16bf a, v16bf b, v8f c) {
    return __builtin_amdgcn_wmma_f32_16x16x32_bf16(
        false, a, false, b, (short)0, c, false, false);
}

// ---------------------------------------------------------------------------
// Pre-pass: convert fp32 weights to bf16 (B^T layout == native row-major).
__global__ __launch_bounds__(256) void convert_weights_kernel(
    const float* __restrict__ qkvw, const float* __restrict__ projw,
    unsigned short* __restrict__ wq, unsigned short* __restrict__ wp) {
    int i = blockIdx.x * 256 + threadIdx.x;
    if (i < 3 * CDIM * CDIM) wq[i] = f2bf(qkvw[i]);   // [288,96]
    if (i < CDIM * CDIM)     wp[i] = f2bf(projw[i]);  // [96,96]
}

// ---------------------------------------------------------------------------
// Main fused kernel: one workgroup == one wave == one window.
__global__ __launch_bounds__(32)
__attribute__((amdgpu_waves_per_eu(4)))      // cap VGPR budget at 256
void swin_window_attn_kernel(
    const float* __restrict__ x,          // [B,112,112,96]
    const float* __restrict__ qkv_bias,   // [288]
    const float* __restrict__ proj_bias,  // [96]
    const float* __restrict__ rpb,        // [3,49,49]
    const float* __restrict__ mask,       // [256,49,49]
    const unsigned short* __restrict__ wqkv,  // bf16 [288,96]
    const unsigned short* __restrict__ wproj, // bf16 [96,96]
    float* __restrict__ out)              // [B,112,112,96]
{
    const int lane = threadIdx.x;
    const int n16  = lane & 15;
    const int half = lane >> 4;

    const int w  = blockIdx.x;
    const int b  = w >> 8;          // 256 windows per image (16x16)
    const int wi = w & 255;
    const int wh = wi >> 4, ww = wi & 15;

    // LDS (ushort units). 61440 bytes total. ATT reuses dead XB region.
    __shared__ __align__(16) unsigned short smem[30720];
    unsigned short* XB  = smem;             // [64,96] bf16 input window
    unsigned short* ATT = smem;             // [64,64] bf16 softmax probs (reuse)
    unsigned short* Q   = smem + 6144;      // [64,96]
    unsigned short* K   = smem + 12288;     // [64,96]
    unsigned short* VT  = smem + 18432;     // [96,64] (v transposed)
    unsigned short* OB  = smem + 24576;     // [64,96] attention output

    // ---- Phase 0: load shifted window, fp32 -> bf16 (vectorized) ----------
    // 64 rows x 24 float4 per row; rows are 384B so float4 loads stay aligned.
#pragma unroll 1
    for (int idx = lane; idx < SPAD * (CDIM / 4); idx += 32) {
        int m = idx / (CDIM / 4), c4 = idx - m * (CDIM / 4);
        u32x2 packed = {0u, 0u};
        if (m < 49) {
            int r  = m / SWIN, cl = m - r * SWIN;
            int gr = (wh * SWIN + r  + SSHIFT) % HW;
            int gc = (ww * SWIN + cl + SSHIFT) % HW;
            const f32x4 f =
                *(const f32x4*)(x + ((size_t)(b * HW + gr) * HW + gc) * CDIM + c4 * 4);
            packed.x = pk2bf(f.x, f.y);
            packed.y = pk2bf(f.z, f.w);
        }
        *(u32x2*)(&XB[m * CDIM + c4 * 4]) = packed;
    }
    asm volatile("s_wait_dscnt 0x0" ::: "memory");

    // ---- Phase 1: QKV = X @ Wqkv^T + b ; scatter into Q / K / V^T ---------
#pragma unroll 1
    for (int mt = 0; mt < 4; ++mt) {
        v16bf a0 = ldA(XB, mt * 16 + n16, CDIM, 0,  half);
        v16bf a1 = ldA(XB, mt * 16 + n16, CDIM, 32, half);
        v16bf a2 = ldA(XB, mt * 16 + n16, CDIM, 64, half);
#pragma unroll 1
        for (int nt = 0; nt < 18; ++nt) {
            const unsigned short* bp = wqkv + (size_t)(nt * 16 + n16) * CDIM;
            v8f acc = {};
            acc = wmma_bf16(a0, ldB(bp, 0,  half), acc);
            acc = wmma_bf16(a1, ldB(bp, 32, half), acc);
            acc = wmma_bf16(a2, ldB(bp, 64, half), acc);
            float bias = qkv_bias[nt * 16 + n16];
            int sec = nt / 6;                  // 0=q 1=k 2=v
            int col = (nt % 6) * 16 + n16;     // head*32 + d in [0,96)
            if (sec == 2) {                    // V^T: consecutive j -> packed b32
#pragma unroll
                for (int j = 0; j < 8; j += 2) {
                    int m = mt * 16 + half * 8 + j;   // even
                    *(unsigned*)(&VT[col * SPAD + m]) =
                        pk2bf(acc[j] + bias, acc[j + 1] + bias);
                }
            } else {
                unsigned short* dst = (sec == 0) ? Q : K;
                float scale = (sec == 0) ? 0.17677669529663687f : 1.0f;
#pragma unroll
                for (int j = 0; j < 8; ++j) {
                    int m = mt * 16 + half * 8 + j;
                    dst[m * CDIM + col] = f2bf((acc[j] + bias) * scale);
                }
            }
        }
    }
    asm volatile("s_wait_dscnt 0x0" ::: "memory");

    // ---- Phase 2: per-head attention --------------------------------------
#pragma unroll 1
    for (int h = 0; h < NHEADS; ++h) {
        // scores = q @ k^T  (K-dim = 32, one WMMA per N-tile), softmax in regs
#pragma unroll 1
        for (int mt = 0; mt < 4; ++mt) {
            v16bf aq = ldA(Q, mt * 16 + n16, CDIM, h * HD, half);
            v8f sc[4];
#pragma unroll
            for (int ntt = 0; ntt < 4; ++ntt) {
                v16bf bk = ldB(K + (size_t)(ntt * 16 + n16) * CDIM + h * HD, 0, half);
                v8f z = {};
                sc[ntt] = wmma_bf16(aq, bk, z);
            }
#pragma unroll
            for (int j = 0; j < 8; ++j) {
                int m = mt * 16 + half * 8 + j;
                bool mok = (m < 49);
                float v0[4];
#pragma unroll
                for (int ntt = 0; ntt < 4; ++ntt) {
                    int t = ntt * 16 + n16;
                    float s = sc[ntt][j];
                    if (mok && t < 49)
                        s += rpb[(h * 49 + m) * 49 + t] +
                             mask[((size_t)wi * 49 + m) * 49 + t];
                    else
                        s = -1e30f;
                    v0[ntt] = s;
                }
                float mx = fmaxf(fmaxf(v0[0], v0[1]), fmaxf(v0[2], v0[3]));
#pragma unroll
                for (int d = 1; d < 16; d <<= 1)
                    mx = fmaxf(mx, __shfl_xor(mx, d, 32));   // rows live per 16-lane half
                float e[4], sum = 0.f;
#pragma unroll
                for (int ntt = 0; ntt < 4; ++ntt) {
                    e[ntt] = (v0[ntt] <= -1e29f) ? 0.f : __expf(v0[ntt] - mx);
                    sum += e[ntt];
                }
#pragma unroll
                for (int d = 1; d < 16; d <<= 1)
                    sum += __shfl_xor(sum, d, 32);
                float rinv = (sum > 0.f) ? 1.0f / sum : 0.f;
#pragma unroll
                for (int ntt = 0; ntt < 4; ++ntt)
                    ATT[m * SPAD + ntt * 16 + n16] = f2bf(e[ntt] * rinv);
            }
        }
        asm volatile("s_wait_dscnt 0x0" ::: "memory");

        // out_h = P @ V  (K-dim = 64 padded tokens; padding probs are 0)
#pragma unroll 1
        for (int mt = 0; mt < 4; ++mt) {
            v16bf p0 = ldA(ATT, mt * 16 + n16, SPAD, 0,  half);
            v16bf p1 = ldA(ATT, mt * 16 + n16, SPAD, 32, half);
#pragma unroll 1
            for (int ntt = 0; ntt < 2; ++ntt) {
                const unsigned short* vp = VT + (size_t)(h * HD + ntt * 16 + n16) * SPAD;
                v8f acc = {};
                acc = wmma_bf16(p0, ldB(vp, 0,  half), acc);
                acc = wmma_bf16(p1, ldB(vp, 32, half), acc);
#pragma unroll
                for (int j = 0; j < 8; ++j) {
                    int m = mt * 16 + half * 8 + j;
                    OB[m * CDIM + h * HD + ntt * 16 + n16] = f2bf(acc[j]);
                }
            }
        }
        asm volatile("s_wait_dscnt 0x0" ::: "memory");
    }

    // ---- Phase 3: proj + bias, store with window-reverse & un-shift -------
#pragma unroll 1
    for (int mt = 0; mt < 4; ++mt) {
        v16bf a0 = ldA(OB, mt * 16 + n16, CDIM, 0,  half);
        v16bf a1 = ldA(OB, mt * 16 + n16, CDIM, 32, half);
        v16bf a2 = ldA(OB, mt * 16 + n16, CDIM, 64, half);
#pragma unroll 1
        for (int nt = 0; nt < 6; ++nt) {
            const unsigned short* bp = wproj + (size_t)(nt * 16 + n16) * CDIM;
            v8f acc = {};
            acc = wmma_bf16(a0, ldB(bp, 0,  half), acc);
            acc = wmma_bf16(a1, ldB(bp, 32, half), acc);
            acc = wmma_bf16(a2, ldB(bp, 64, half), acc);
            float bias = proj_bias[nt * 16 + n16];
            int c = nt * 16 + n16;
#pragma unroll
            for (int j = 0; j < 8; ++j) {
                int m = mt * 16 + half * 8 + j;
                if (m < 49) {
                    int r  = m / SWIN, cl = m - r * SWIN;
                    int gr = (wh * SWIN + r  + SSHIFT) % HW;
                    int gc = (ww * SWIN + cl + SSHIFT) % HW;
                    out[((size_t)(b * HW + gr) * HW + gc) * CDIM + c] = acc[j] + bias;
                }
            }
        }
    }
}

// ---------------------------------------------------------------------------
extern "C" void kernel_launch(void* const* d_in, const int* in_sizes, int n_in,
                              void* d_out, int out_size, void* d_ws, size_t ws_size,
                              hipStream_t stream) {
    (void)n_in; (void)out_size; (void)ws_size;
    const float* x         = (const float*)d_in[0];
    const float* qkv_w     = (const float*)d_in[1];
    const float* qkv_b     = (const float*)d_in[2];
    const float* proj_w    = (const float*)d_in[3];
    const float* proj_b    = (const float*)d_in[4];
    const float* rpb       = (const float*)d_in[5];
    const float* attn_mask = (const float*)d_in[6];
    float* out = (float*)d_out;

    // Workspace: bf16 weights. wqkv [288*96], wproj [96*96].
    unsigned short* wqkv  = (unsigned short*)d_ws;
    unsigned short* wproj = (unsigned short*)((char*)d_ws + 3 * CDIM * CDIM * sizeof(unsigned short));

    int nconv = 3 * CDIM * CDIM;   // 27648
    convert_weights_kernel<<<(nconv + 255) / 256, 256, 0, stream>>>(
        qkv_w, proj_w, wqkv, wproj);

    int B = in_sizes[0] / (HW * HW * CDIM);        // 64
    int nwin = B * (HW / SWIN) * (HW / SWIN);      // 16384 windows
    swin_window_attn_kernel<<<nwin, 32, 0, stream>>>(
        x, qkv_b, proj_b, rpb, attn_mask, wqkv, wproj, out);
}
